// FastAttention_70574902608390
// MI455X (gfx1250) — matrix operand order repaired and checked by reading, not compiled, not run on hardware
//
#include <hip/hip_runtime.h>

// ---------------- problem constants (from reference) ----------------
constexpr int kB = 2, kT = 4096, kC = 1024, kH = 16, kM = 64, kD = 64;
constexpr int kChunk = 128, kNCH = kT / kChunk;   // 32 chunks
#define DELTA_EPS 1e-6f

typedef _Float16 h16;
typedef __attribute__((ext_vector_type(16))) _Float16 v16h;
typedef __attribute__((ext_vector_type(8)))  float    v8f;

// ---------------- async memory->LDS path (CDNA5 GLOBAL_LOAD_ASYNC_TO_LDS_B128) ----------------
#define AS1 __attribute__((address_space(1)))
#define AS3 __attribute__((address_space(3)))

#if defined(__AMDGCN__) && __has_builtin(__builtin_amdgcn_global_load_async_to_lds_b128)
#define HAVE_ASYNC_LDS 1
#else
#define HAVE_ASYNC_LDS 0
#endif

// copy 32 bytes (16 f16) global -> LDS
__device__ __forceinline__ void cp32(h16* dst, const h16* src) {
#if HAVE_ASYNC_LDS
  typedef __attribute__((__vector_size__(4 * sizeof(int)))) int i4vec;
  __builtin_amdgcn_global_load_async_to_lds_b128((AS1 i4vec*)src, (AS3 i4vec*)dst, 0, 0);
  __builtin_amdgcn_global_load_async_to_lds_b128((AS1 i4vec*)(src + 8), (AS3 i4vec*)(dst + 8), 0, 0);
#else
  ((uint4*)dst)[0] = ((const uint4*)src)[0];
  ((uint4*)dst)[1] = ((const uint4*)src)[1];
#endif
}

// wait for this wave's outstanding async loads (ASYNCcnt==0); call before __syncthreads()
__device__ __forceinline__ void async_fence() {
#if HAVE_ASYNC_LDS
#if __has_builtin(__builtin_amdgcn_s_wait_asynccnt)
  __builtin_amdgcn_s_wait_asynccnt(0);
#else
  asm volatile("s_wait_asynccnt 0" ::: "memory");
#endif
#endif
}

// ---------------- WMMA helpers (wave32, 16x16x32 f16 -> f32 acc) ----------------
__device__ __forceinline__ v8f wmma16(v16h a, v16h b, v8f c) {
  // 8 args: (neg_a, A, neg_b, B, c_mod, C, reuse_a, reuse_b)
  return __builtin_amdgcn_wmma_f32_16x16x32_f16(false, a, false, b, (short)0, c, false, false);
}

// A fragment, row-major source S[m][k] (stride ld elems).
__device__ __forceinline__ v16h packA_row(const h16* S, int ld, int row0, int k0, int lane) {
  int m  = row0 + (lane & 15);
  int kh = (lane >> 4) << 4;
  const h16* p = S + (size_t)m * ld + k0 + kh;
  union { uint4 u[2]; v16h v; } U;
  U.u[0] = *(const uint4*)p;
  U.u[1] = *(const uint4*)(p + 8);
  return U.v;
}

// B fragment, "weight" layout S[n][k] (stride ld elems).
__device__ __forceinline__ v16h packB_row(const h16* S, int ld, int n0, int k0, int lane) {
  int n  = n0 + (lane & 15);
  int kb = (lane >> 4) << 4;
  const h16* p = S + (size_t)n * ld + k0 + kb;
  union { uint4 u[2]; v16h v; } U;
  U.u[0] = *(const uint4*)p;
  U.u[1] = *(const uint4*)(p + 8);
  return U.v;
}

// B fragment from an f32 [n][k] source (converted per element).
__device__ __forceinline__ v16h packB_f32(const float* S, int ld, int n0, int k0, int lane) {
  int n  = n0 + (lane & 15);
  int kb = k0 + ((lane >> 4) << 4);
  const float* p = S + (size_t)n * ld + kb;
  v16h r;
#pragma unroll
  for (int e = 0; e < 16; ++e) r[e] = (h16)p[e];
  return r;
}

// B fragment where storage is transposed: element (n,k) lives at S[k*ld + n]
__device__ __forceinline__ v16h packB_T(const h16* S, int ld, int n0, int k0, int lane) {
  int n  = n0 + (lane & 15);
  int kb = k0 + ((lane >> 4) << 4);
  v16h r;
#pragma unroll
  for (int e = 0; e < 16; ++e) r[e] = S[(size_t)(kb + e) * ld + n];
  return r;
}

// A fragment where storage is transposed: element (m,k) lives at S[k*ld + m]
__device__ __forceinline__ v16h packA_T(const h16* S, int ld, int row0, int k0, int lane) {
  int m  = row0 + (lane & 15);
  int kh = k0 + ((lane >> 4) << 4);
  v16h r;
#pragma unroll
  for (int e = 0; e < 8; ++e) {
    r[e]     = S[(size_t)(kh + e) * ld + m];
    r[e + 8] = S[(size_t)(kh + 8 + e) * ld + m];
  }
  return r;
}

// ---------------- conversion kernels ----------------
__global__ void k_f32_to_f16(const float* __restrict__ src, h16* __restrict__ dst, int n) {
  int i = blockIdx.x * blockDim.x + threadIdx.x;
  int stride = gridDim.x * blockDim.x;
  for (; i < n; i += stride) dst[i] = (h16)src[i];
}

// proj (H,D,M) f32 -> projT (H,M,D) f16
__global__ void k_projT(const float* __restrict__ proj, h16* __restrict__ projT) {
  int i = blockIdx.x * blockDim.x + threadIdx.x;
  if (i >= kH * kM * kD) return;
  int h = i / (kM * kD);
  int rem = i % (kM * kD);
  int m = rem / kD, d = rem % kD;
  projT[i] = (h16)proj[((size_t)h * kD + d) * kM + m];
}

// ---------------- tiled WMMA GEMM: out[M,N] = A[M,K] @ W[N,K]^T + bias ----------------
// block tile 256x128, 8 waves, wave tile 64x64 (16 WMMA / K-step, ~1 ds_load_b128 per WMMA)
constexpr int GM = 256, GN = 128, GK = 32, GLD = 40;

__global__ __launch_bounds__(256) void k_gemm(const h16* __restrict__ A, const h16* __restrict__ W,
                                              const float* __restrict__ bias,
                                              float* __restrict__ outF, h16* __restrict__ outH,
                                              int Mtot, int Ntot, int Ktot) {
  __shared__ h16 sA[GM * GLD];   // 20480 B
  __shared__ h16 sB[GN * GLD];   // 10240 B
  int tid = threadIdx.x, lane = tid & 31, wv = tid >> 5;
  int m0 = blockIdx.y * GM, n0 = blockIdx.x * GN;
  int wm = (wv & 3) * 64, wn = (wv >> 2) * 64;
  v8f acc[4][4] = {};

  int ar = tid >> 1, aseg = (tid & 1) * 16;   // A: rows ar and ar+128, 16 elems (32B) each
  int br = tid >> 1, bseg = (tid & 1) * 16;   // B: row br, 16 elems

  for (int kt = 0; kt < Ktot; kt += GK) {
    __syncthreads();
    cp32(sA + ar * GLD + aseg,          A + (size_t)(m0 + ar) * Ktot + kt + aseg);
    cp32(sA + (ar + 128) * GLD + aseg,  A + (size_t)(m0 + ar + 128) * Ktot + kt + aseg);
    cp32(sB + br * GLD + bseg,          W + (size_t)(n0 + br) * Ktot + kt + bseg);
    if (kt + GK < Ktot) {  // hint next K tiles toward cache
      __builtin_prefetch(A + (size_t)(m0 + ar) * Ktot + kt + GK + aseg, 0, 1);
      __builtin_prefetch(W + (size_t)(n0 + br) * Ktot + kt + GK + bseg, 0, 1);
    }
    async_fence();
    __syncthreads();
    v16h af[4], bf[4];
#pragma unroll
    for (int i = 0; i < 4; ++i) af[i] = packA_row(sA, GLD, wm + i * 16, 0, lane);
#pragma unroll
    for (int j = 0; j < 4; ++j) bf[j] = packB_row(sB, GLD, wn + j * 16, 0, lane);
#pragma unroll
    for (int i = 0; i < 4; ++i)
#pragma unroll
      for (int j = 0; j < 4; ++j) acc[i][j] = wmma16(af[i], bf[j], acc[i][j]);
  }

#pragma unroll
  for (int i = 0; i < 4; ++i)
#pragma unroll
    for (int j = 0; j < 4; ++j) {
      int nn = n0 + wn + j * 16 + (lane & 15);
      float bb = bias[nn];
#pragma unroll
      for (int r = 0; r < 8; ++r) {
        int mm = m0 + wm + i * 16 + r + ((lane >> 4) << 3);
        float v = acc[i][j][r] + bb;
        if (outF) outF[(size_t)mm * Ntot + nn] = v;
        if (outH) outH[(size_t)mm * Ntot + nn] = (h16)v;
      }
    }
}

// ---------------- shared attention building block: relu(sIn @ sPJ^T) ----------------
__device__ __forceinline__ void proj_relu(const h16* sIn, const h16* sPJ, h16* sOut, int wv, int lane) {
  v8f acc0 = {}, acc1 = {}, acc2 = {}, acc3 = {};
#pragma unroll
  for (int kk = 0; kk < 2; ++kk) {
    v16h a = packA_row(sIn, 64, wv * 16, kk * 32, lane);
    acc0 = wmma16(a, packB_row(sPJ, 64, 0, kk * 32, lane), acc0);
    acc1 = wmma16(a, packB_row(sPJ, 64, 16, kk * 32, lane), acc1);
    acc2 = wmma16(a, packB_row(sPJ, 64, 32, kk * 32, lane), acc2);
    acc3 = wmma16(a, packB_row(sPJ, 64, 48, kk * 32, lane), acc3);
  }
  v8f* accs[4] = {&acc0, &acc1, &acc2, &acc3};
#pragma unroll
  for (int j = 0; j < 4; ++j)
#pragma unroll
    for (int r = 0; r < 8; ++r) {
      int tl = wv * 16 + r + ((lane >> 4) << 3);
      int mm = j * 16 + (lane & 15);
      float v = (*accs[j])[r];
      sOut[tl * 64 + mm] = (h16)(v > 0.f ? v : 0.f);
    }
}

// ---------------- pass A: per-chunk state increments ----------------
__global__ __launch_bounds__(256) void k_chunk_state(const h16* __restrict__ qkv,
                                                     const h16* __restrict__ projT,
                                                     float* __restrict__ Skv, float* __restrict__ Sk) {
  __shared__ h16 sPJ[64 * 64];
  __shared__ h16 sX[128 * 64];
  __shared__ h16 sKP[128 * 64];
  __shared__ h16 sV[128 * 64];
  int id = blockIdx.x;                     // bh * NCH + ch
  int ch = id % kNCH, bh = id / kNCH;
  int b = bh / kH, h = bh % kH;
  int tid = threadIdx.x, lane = tid & 31, wv = tid >> 5;
  const h16* base = qkv + (size_t)b * kT * 3 * kC + h * kD;
  const h16* kbp = base + kC;
  const h16* vbp = base + 2 * kC;
  size_t t0 = (size_t)ch * kChunk;
  int rowt = tid >> 1, segt = (tid & 1) * 32;

  cp32(sPJ + tid * 16, projT + (size_t)h * kM * kD + tid * 16);
  {
    const h16* s = kbp + (t0 + rowt) * 3 * kC + segt;
    h16* d = sX + rowt * 64 + segt;
    cp32(d, s); cp32(d + 16, s + 16);
  }
  {
    const h16* s = vbp + (t0 + rowt) * 3 * kC + segt;
    h16* d = sV + rowt * 64 + segt;
    cp32(d, s); cp32(d + 16, s + 16);
  }
  async_fence();
  __syncthreads();
  proj_relu(sX, sPJ, sKP, wv, lane);
  __syncthreads();

  // Skv = v^T @ kp : GEMM M=d(64), N=m(64), K=t(128); wave: 16 rows x 32 cols
  v8f kv0 = {}, kv1 = {};
  int rm = (wv & 3) * 16, cn = (wv >> 2) * 32;
#pragma unroll
  for (int kk = 0; kk < 4; ++kk) {
    v16h a = packA_T(sV, 64, rm, kk * 32, lane);
    kv0 = wmma16(a, packB_T(sKP, 64, cn, kk * 32, lane), kv0);
    kv1 = wmma16(a, packB_T(sKP, 64, cn + 16, kk * 32, lane), kv1);
  }
  float* dst = Skv + (size_t)id * (kM * kD);
  v8f* kvs[2] = {&kv0, &kv1};
#pragma unroll
  for (int j = 0; j < 2; ++j)
#pragma unroll
    for (int r = 0; r < 8; ++r) {
      int d = rm + r + ((lane >> 4) << 3);
      int m = cn + j * 16 + (lane & 15);
      dst[d * kM + m] = (*kvs[j])[r];
    }
  if (tid < kM) {
    float s = 0.f;
#pragma unroll 8
    for (int t = 0; t < kChunk; ++t) s += (float)sKP[t * 64 + tid];
    Sk[(size_t)id * kM + tid] = s;
  }
}

// ---------------- pass B: exclusive prefix over chunks ----------------
__global__ __launch_bounds__(256) void k_prefix(const float* __restrict__ Skv, const float* __restrict__ Sk,
                                                float* __restrict__ kvpre, float* __restrict__ kpre) {
  int bh = blockIdx.x, tid = threadIdx.x;
  float run[16];
#pragma unroll
  for (int i = 0; i < 16; ++i) run[i] = 0.f;
  for (int ch = 0; ch < kNCH; ++ch) {
    size_t base = ((size_t)bh * kNCH + ch) * (kM * kD);
#pragma unroll
    for (int i = 0; i < 16; ++i) {
      int p = tid * 16 + i;
      kvpre[base + p] = run[i];
      run[i] += Skv[base + p];
    }
  }
  if (tid < kM) {
    float r = 0.f;
    for (int ch = 0; ch < kNCH; ++ch) {
      size_t base = ((size_t)bh * kNCH + ch) * kM;
      kpre[base + tid] = r;
      r += Sk[base + tid];
    }
  }
}

// ---------------- pass C: per-chunk attention output ----------------
constexpr int A_PJ = 0;              // 64*64*2   = 8192
constexpr int A_QP = 8192;           // 128*64*2  = 16384
constexpr int A_KP = 24576;          // 16384
constexpr int A_V  = 40960;          // 16384
constexpr int A_NR = 57344;          // 128*4     = 512
constexpr int A_X  = 57856;          // staging 16384, aliased by AW 128*128*2 = 32768
constexpr int ATTN_SMEM = 90624;

__global__ __launch_bounds__(256) void k_attn_out(const h16* __restrict__ qkv, const h16* __restrict__ projT,
                                                  const float* __restrict__ kvpre, const float* __restrict__ kpre,
                                                  h16* __restrict__ att) {
  extern __shared__ char smem[];
  h16*   sPJ = (h16*)(smem + A_PJ);
  h16*   sQP = (h16*)(smem + A_QP);
  h16*   sKP = (h16*)(smem + A_KP);
  h16*   sV  = (h16*)(smem + A_V);
  float* sNR = (float*)(smem + A_NR);
  h16*   sX  = (h16*)(smem + A_X);
  h16*   sAW = sX;  // [128][128], overlaps dead staging buffer

  int id = blockIdx.x;
  int ch = id % kNCH, bh = id / kNCH;
  int b = bh / kH, h = bh % kH;
  int tid = threadIdx.x, lane = tid & 31, wv = tid >> 5;
  const h16* qbp = qkv + (size_t)b * kT * 3 * kC + h * kD;
  const h16* kbp = qbp + kC;
  const h16* vbp = qbp + 2 * kC;
  size_t t0 = (size_t)ch * kChunk;
  int rowt = tid >> 1, segt = (tid & 1) * 32;

  cp32(sPJ + tid * 16, projT + (size_t)h * kM * kD + tid * 16);
  {
    const h16* s = qbp + (t0 + rowt) * 3 * kC + segt;
    h16* d = sX + rowt * 64 + segt;
    cp32(d, s); cp32(d + 16, s + 16);
  }
  async_fence();
  __syncthreads();
  proj_relu(sX, sPJ, sQP, wv, lane);
  __syncthreads();
  {
    const h16* s = kbp + (t0 + rowt) * 3 * kC + segt;
    h16* d = sX + rowt * 64 + segt;
    cp32(d, s); cp32(d + 16, s + 16);
  }
  async_fence();
  __syncthreads();
  proj_relu(sX, sPJ, sKP, wv, lane);
  {
    const h16* s = vbp + (t0 + rowt) * 3 * kC + segt;
    h16* d = sV + rowt * 64 + segt;
    cp32(d, s); cp32(d + 16, s + 16);
  }
  async_fence();
  __syncthreads();

  // ---- ctx_hist: qp @ KVpre^T   (KVpre stored [d][m] => B[n=d][k=m] rows contiguous)
  v8f ctx0 = {}, ctx1 = {}, ctx2 = {}, ctx3 = {};
  const float* kvb = kvpre + (size_t)id * (kM * kD);
#pragma unroll
  for (int kk = 0; kk < 2; ++kk) {
    v16h a = packA_row(sQP, 64, wv * 16, kk * 32, lane);
    ctx0 = wmma16(a, packB_f32(kvb, kM, 0, kk * 32, lane), ctx0);
    ctx1 = wmma16(a, packB_f32(kvb, kM, 16, kk * 32, lane), ctx1);
    ctx2 = wmma16(a, packB_f32(kvb, kM, 32, kk * 32, lane), ctx2);
    ctx3 = wmma16(a, packB_f32(kvb, kM, 48, kk * 32, lane), ctx3);
  }

  // ---- aw = qp @ kp^T with causal mask -> sAW (f16)
  {
    v8f aw0 = {}, aw1 = {}, aw2 = {}, aw3 = {}, aw4 = {}, aw5 = {}, aw6 = {}, aw7 = {};
#pragma unroll
    for (int kk = 0; kk < 2; ++kk) {
      v16h a = packA_row(sQP, 64, wv * 16, kk * 32, lane);
      aw0 = wmma16(a, packB_row(sKP, 64, 0, kk * 32, lane), aw0);
      aw1 = wmma16(a, packB_row(sKP, 64, 16, kk * 32, lane), aw1);
      aw2 = wmma16(a, packB_row(sKP, 64, 32, kk * 32, lane), aw2);
      aw3 = wmma16(a, packB_row(sKP, 64, 48, kk * 32, lane), aw3);
      aw4 = wmma16(a, packB_row(sKP, 64, 64, kk * 32, lane), aw4);
      aw5 = wmma16(a, packB_row(sKP, 64, 80, kk * 32, lane), aw5);
      aw6 = wmma16(a, packB_row(sKP, 64, 96, kk * 32, lane), aw6);
      aw7 = wmma16(a, packB_row(sKP, 64, 112, kk * 32, lane), aw7);
    }
    v8f* aws[8] = {&aw0, &aw1, &aw2, &aw3, &aw4, &aw5, &aw6, &aw7};
#pragma unroll
    for (int j = 0; j < 8; ++j)
#pragma unroll
      for (int r = 0; r < 8; ++r) {
        int tl = wv * 16 + r + ((lane >> 4) << 3);
        int s = j * 16 + (lane & 15);
        float v = (s <= tl) ? (*aws[j])[r] : 0.f;
        sAW[tl * 128 + s] = (h16)v;
      }
  }
  __syncthreads();

  // ---- norms (intra row-sum + history dot with kpre); VALU only, uniform flow around WMMA
  if (tid < kChunk) {
    float nrm = DELTA_EPS;
#pragma unroll 8
    for (int s = 0; s < kChunk; ++s) nrm += (float)sAW[tid * 128 + s];
    const float* kp_pre = kpre + (size_t)id * kM;
#pragma unroll 8
    for (int m = 0; m < kM; ++m) nrm += (float)sQP[tid * 64 + m] * kp_pre[m];
    sNR[tid] = nrm;
  }

  // ---- ctx_intra: aw @ v   (B[n=d][k=s] gathered from sV[s][d])
#pragma unroll
  for (int kk = 0; kk < 4; ++kk) {
    v16h a = packA_row(sAW, 128, wv * 16, kk * 32, lane);
    ctx0 = wmma16(a, packB_T(sV, 64, 0, kk * 32, lane), ctx0);
    ctx1 = wmma16(a, packB_T(sV, 64, 16, kk * 32, lane), ctx1);
    ctx2 = wmma16(a, packB_T(sV, 64, 32, kk * 32, lane), ctx2);
    ctx3 = wmma16(a, packB_T(sV, 64, 48, kk * 32, lane), ctx3);
  }
  __syncthreads();

  // ---- normalize + store att (B,T,H*D) f16
  v8f* ctxs[4] = {&ctx0, &ctx1, &ctx2, &ctx3};
#pragma unroll
  for (int j = 0; j < 4; ++j)
#pragma unroll
    for (int r = 0; r < 8; ++r) {
      int tl = wv * 16 + r + ((lane >> 4) << 3);
      int d = j * 16 + (lane & 15);
      float o = (*ctxs[j])[r] / sNR[tl];
      att[((size_t)b * kT + t0 + tl) * kC + h * kD + d] = (h16)o;
    }
}

// ---------------- host launcher ----------------
extern "C" void kernel_launch(void* const* d_in, const int* in_sizes, int n_in,
                              void* d_out, int out_size, void* d_ws, size_t ws_size,
                              hipStream_t stream) {
  (void)in_sizes; (void)n_in; (void)out_size; (void)ws_size;
  const float* x    = (const float*)d_in[0];
  const float* Wqkv = (const float*)d_in[1];
  const float* bqkv = (const float*)d_in[2];
  const float* Wout = (const float*)d_in[3];
  const float* bout = (const float*)d_in[4];
  const float* proj = (const float*)d_in[5];

  char* ws = (char*)d_ws;
  size_t off = 0;
  auto take = [&](size_t bytes) -> char* {
    char* p = ws + off;
    off += (bytes + 255) & ~(size_t)255;
    return p;
  };
  h16*   xB     = (h16*)take((size_t)kB * kT * kC * 2);
  h16*   qkvB   = (h16*)take((size_t)kB * kT * 3 * kC * 2);
  h16*   attB   = (h16*)take((size_t)kB * kT * kC * 2);
  h16*   WqkvB  = (h16*)take((size_t)3 * kC * kC * 2);
  h16*   WoutB  = (h16*)take((size_t)kC * kC * 2);
  h16*   projTB = (h16*)take((size_t)kH * kM * kD * 2);
  float* Skv    = (float*)take((size_t)kB * kH * kNCH * kM * kD * 4);
  float* Sk     = (float*)take((size_t)kB * kH * kNCH * kM * 4);
  float* kvpre  = (float*)take((size_t)kB * kH * kNCH * kM * kD * 4);
  float* kpre   = (float*)take((size_t)kB * kH * kNCH * kM * 4);

  (void)hipFuncSetAttribute(reinterpret_cast<const void*>(k_attn_out),
                            hipFuncAttributeMaxDynamicSharedMemorySize, ATTN_SMEM);

  // 1) precision conversion
  k_f32_to_f16<<<2048, 256, 0, stream>>>(x, xB, kB * kT * kC);
  k_f32_to_f16<<<2048, 256, 0, stream>>>(Wqkv, WqkvB, 3 * kC * kC);
  k_f32_to_f16<<<1024, 256, 0, stream>>>(Wout, WoutB, kC * kC);
  k_projT<<<(kH * kM * kD + 255) / 256, 256, 0, stream>>>(proj, projTB);

  // 2) QKV GEMM: (8192 x 3072) = x @ Wqkv^T + bqkv
  k_gemm<<<dim3(3 * kC / GN, kB * kT / GM), 256, 0, stream>>>(
      xB, WqkvB, bqkv, nullptr, qkvB, kB * kT, 3 * kC, kC);

  // 3) chunked linear attention, scan parallelized via prefix
  k_chunk_state<<<kB * kH * kNCH, 256, 0, stream>>>(qkvB, projTB, Skv, Sk);
  k_prefix<<<kB * kH, 256, 0, stream>>>(Skv, Sk, kvpre, kpre);
  k_attn_out<<<kB * kH * kNCH, 256, ATTN_SMEM, stream>>>(qkvB, projTB, kvpre, kpre, attB);

  // 4) output GEMM: (8192 x 1024) = att @ Wout^T + bout -> fp32 d_out
  k_gemm<<<dim3(kC / GN, kB * kT / GM), 256, 0, stream>>>(
      attB, WoutB, bout, (float*)d_out, nullptr, kB * kT, kC, kC);
}
